// MultiScalePatchTokenizer_19224273617477
// MI455X (gfx1250) — matrix operand order repaired and checked by reading, not compiled, not run on hardware
//
#include <hip/hip_runtime.h>
#include <hip/hip_bf16.h>
#include <math.h>

typedef __attribute__((ext_vector_type(16))) _Float16 v16h;
typedef __attribute__((ext_vector_type(8)))  float    v8f;

#define B_    16
#define N_    4096
#define S_    512
#define STEM  128
#define TOK   256
#define DIN   155
#define DINP  160
#define DTOK  992
#define EPS_  1e-5f

// ---------------- helpers ----------------

__device__ __forceinline__ float gelu_exact(float x) {
    return 0.5f * x * (1.0f + erff(x * 0.70710678118654752f));
}

// A-fragment (16x32 f16), contiguous K order matching the swizzled B:
// lanes 0-15: row=lane, elems 0..15 -> K=k0..k0+15
// lanes 16-31: row=lane-16, elems 0..15 -> K=k0+16..k0+31
__device__ __forceinline__ v16h load_a_frag(const _Float16* base, int stride, int k0) {
    int lane = threadIdx.x & 31;
    int row  = lane & 15;
    int off  = (lane >> 4) * 16;
    const _Float16* p = base + (long)row * stride + k0;
    v16h a;
#pragma unroll
    for (int e = 0; e < 8; ++e) a[e] = p[off + e];
#pragma unroll
    for (int e = 0; e < 8; ++e) a[8 + e] = p[off + 8 + e];
    return a;
}

// B-fragment from pre-swizzled (fragment-major) weights: each lane's 16 f16
// elements are contiguous -> one 32B load (2x global_load_b128).
__device__ __forceinline__ v16h load_b_frag_sw(const _Float16* Wsw, int frag) {
    int lane = threadIdx.x & 31;
    const v16h* p = (const v16h*)(Wsw + ((long)frag * 32 + lane) * 16);
    return *p;
}

__device__ __forceinline__ float wave_sum(float v) {
#pragma unroll
    for (int m = 16; m >= 1; m >>= 1) v += __shfl_xor(v, m, 32);
    return v;
}

// ---------------- weight conversion: fp32 -> f16, K pad, B-fragment swizzle ----
// dst element layout: frag = kt*16 + nt ; within frag: lane (0..31), e (0..15)
//   maps to src[k = kt*32 + (lane>>4)*16 + e][n = nt*16 + (lane&15)]
// so a B fragment is one contiguous 32B run per lane.

__global__ void convert_swizzle_kernel(const float* __restrict__ src, _Float16* __restrict__ dst,
                                       int ksrc, int kpad, int batch) {
    long per   = (long)kpad * TOK;            // elements per matrix
    long total = (long)batch * per;
    for (long i = (long)blockIdx.x * blockDim.x + threadIdx.x; i < total;
         i += (long)gridDim.x * blockDim.x) {
        int  s    = (int)(i / per);
        long rem  = i % per;
        int  frag = (int)(rem >> 9);          // /512
        int  wi   = (int)(rem & 511);
        int  lane = wi >> 4;
        int  e    = wi & 15;
        int  kt   = frag >> 4;
        int  nt   = frag & 15;
        int  k    = kt * 32 + (lane >> 4) * 16 + e;
        int  n    = nt * 16 + (lane & 15);
        float v = (k < ksrc) ? src[((long)s * ksrc + k) * TOK + n] : 0.0f;
        dst[i] = (_Float16)v;
    }
}

// ---------------- farthest point sampling ----------------

__global__ void __launch_bounds__(256) fps_kernel(const float* __restrict__ xyz,
                                                  int* __restrict__ center_idx) {
    __shared__ float px[N_], py[N_], pz[N_];
    __shared__ float rv[256];
    __shared__ int   ri[256];
    __shared__ float cc[3];
    int b = blockIdx.x, tid = threadIdx.x;
    const float* base = xyz + (long)b * N_ * 3;
    for (int i = tid; i < N_; i += 256) {
        px[i] = base[i * 3 + 0];
        py[i] = base[i * 3 + 1];
        pz[i] = base[i * 3 + 2];
    }
    float dist[16];
#pragma unroll
    for (int i = 0; i < 16; ++i) dist[i] = 1e10f;
    __syncthreads();
    if (tid == 0) { cc[0] = px[0]; cc[1] = py[0]; cc[2] = pz[0]; }
    int far = 0;
    for (int it = 0; it < S_; ++it) {
        if (tid == 0) center_idx[b * S_ + it] = far;
        __syncthreads();
        float cx = cc[0], cy = cc[1], cz = cc[2];
        float bv = -1.0f;
        int   bi = 0;
#pragma unroll
        for (int i = 0; i < 16; ++i) {
            int   p  = i * 256 + tid;
            float dx = px[p] - cx, dy = py[p] - cy, dz = pz[p] - cz;
            float d  = dx * dx + dy * dy + dz * dz;
            float dm = fminf(dist[i], d);
            dist[i]  = dm;
            if (dm > bv) { bv = dm; bi = p; }
        }
        rv[tid] = bv; ri[tid] = bi;
        __syncthreads();
        for (int s2 = 128; s2 > 0; s2 >>= 1) {
            if (tid < s2) {
                float ov = rv[tid + s2];
                int   oi = ri[tid + s2];
                if (ov > rv[tid] || (ov == rv[tid] && oi < ri[tid])) { rv[tid] = ov; ri[tid] = oi; }
            }
            __syncthreads();
        }
        far = ri[0];
        if (tid == 0) { cc[0] = px[far]; cc[1] = py[far]; cc[2] = pz[far]; }
        __syncthreads();
    }
}

// ---------------- gather centers + center PE ----------------

__global__ void gather_centers_kernel(const float* __restrict__ xyz,
                                      const float* __restrict__ feat,
                                      const int* __restrict__ center_idx,
                                      float* __restrict__ patch_center,
                                      _Float16* __restrict__ tok_in,
                                      float* __restrict__ out_pc) {
    int g = blockIdx.x * blockDim.x + threadIdx.x;
    if (g >= B_ * S_) return;
    int b   = g / S_;
    int idx = center_idx[g];
    const float* p = xyz + ((long)b * N_ + idx) * 3;
    float c0 = p[0], c1 = p[1], c2 = p[2];
    patch_center[g * 3 + 0] = c0; patch_center[g * 3 + 1] = c1; patch_center[g * 3 + 2] = c2;
    out_pc[g * 3 + 0] = c0;       out_pc[g * 3 + 1] = c1;       out_pc[g * 3 + 2] = c2;
    const float* f = feat + ((long)b * N_ + idx) * STEM;
    _Float16* t = tok_in + (long)g * DTOK;
    for (int c = 0; c < STEM; ++c) t[c] = (_Float16)f[c];
    float coord[3] = {c0, c1, c2};
    const float lg = logf(10000.0f);
    for (int ax = 0; ax < 3; ++ax) {
        for (int i = 0; i < 16; ++i) {
            float fr = expf(-lg * (float)i / 15.0f);
            float a  = coord[ax] * fr;
            t[896 + ax * 32 + i]      = (_Float16)sinf(a);
            t[896 + ax * 32 + 16 + i] = (_Float16)cosf(a);
        }
    }
}

// ---------------- ball query: one wave32 per center ----------------

__global__ void __launch_bounds__(256) ball_query_kernel(const float* __restrict__ xyz,
                                                         const float* __restrict__ patch_center,
                                                         int* __restrict__ nbr) {
    const float RAD[3] = {0.1f, 0.2f, 0.4f};
    const int   KK[3]  = {16, 32, 64};
    int scale = blockIdx.y;
    int w     = threadIdx.x >> 5;
    int lane  = threadIdx.x & 31;
    int cg    = blockIdx.x * 8 + w;      // global center id, 0..8191
    int b     = cg / S_;
    float r2  = RAD[scale] * RAD[scale];
    int   k   = KK[scale];
    float cx = patch_center[cg * 3 + 0];
    float cy = patch_center[cg * 3 + 1];
    float cz = patch_center[cg * 3 + 2];
    const float* base = xyz + (long)b * N_ * 3;
    int* out = nbr + ((long)scale * B_ * S_ + cg) * 64;
    int cnt = 0;
    for (int j0 = 0; j0 < N_; j0 += 32) {
        int   p  = j0 + lane;
        float dx = base[p * 3 + 0] - cx;
        float dy = base[p * 3 + 1] - cy;
        float dz = base[p * 3 + 2] - cz;
        float d2 = dx * dx + dy * dy + dz * dz;
        bool in  = d2 <= r2;
        unsigned mask = (unsigned)__ballot(in);
        int slot = cnt + __popc(mask & ((1u << lane) - 1u));
        if (in && slot < k) out[slot] = p;
        cnt += __popc(mask);
        if (cnt >= k) break;
    }
    if (cnt < k) {
        int first = (cnt > 0) ? out[0] : 0;
        for (int s2 = cnt + lane; s2 < k; s2 += 32) out[s2] = first;
    }
}

// ---------------- per-scale group MLP with WMMA + max pool ----------------

__global__ void __launch_bounds__(256) group_mlp_kernel(
    const float* __restrict__ xyz, const float* __restrict__ feat,
    const float* __restrict__ patch_center, const int* __restrict__ nbr,
    const float* __restrict__ sb1, const float* __restrict__ sg1, const float* __restrict__ sbe1,
    const float* __restrict__ sb2, const float* __restrict__ sg2, const float* __restrict__ sbe2,
    const _Float16* __restrict__ sW1sw, const _Float16* __restrict__ sW2sw,
    _Float16* __restrict__ tok_in) {
    const int KK[3] = {16, 32, 64};
    __shared__ float b1s[TOK], g1s[TOK], be1s[TOK], b2s[TOK], g2s[TOK], be2s[TOK];
    __shared__ _Float16 ginh[16][DINP];
    __shared__ float    raw[16][TOK];
    __shared__ _Float16 h1[16][TOK];
    __shared__ float    pooled[TOK];
    __shared__ float    relc[16][3];
    __shared__ int      nidxs[16];

    int scale = blockIdx.y;
    int cg    = blockIdx.x;
    int b     = cg / S_;
    int tid   = threadIdx.x;
    int w     = tid >> 5;
    int lane  = tid & 31;
    int k     = KK[scale];

    b1s[tid] = sb1[scale * TOK + tid];  g1s[tid] = sg1[scale * TOK + tid];
    be1s[tid] = sbe1[scale * TOK + tid];
    b2s[tid] = sb2[scale * TOK + tid];  g2s[tid] = sg2[scale * TOK + tid];
    be2s[tid] = sbe2[scale * TOK + tid];
    pooled[tid] = -3.0e38f;

    float cx = patch_center[cg * 3 + 0];
    float cy = patch_center[cg * 3 + 1];
    float cz = patch_center[cg * 3 + 2];
    const int*      nb = nbr + ((long)scale * B_ * S_ + cg) * 64;
    const _Float16* W1 = sW1sw + (long)scale * DINP * TOK;   // swizzled, same total size
    const _Float16* W2 = sW2sw + (long)scale * TOK * TOK;
    const float lg = logf(10000.0f);

    for (int m0 = 0; m0 < k; m0 += 16) {
        __syncthreads();
        if (tid < 48) {
            int r = tid / 3, ax = tid % 3;
            int nidx = nb[m0 + r];
            if (ax == 0) nidxs[r] = nidx;
            const float* p = xyz + ((long)b * N_ + nidx) * 3;
            float cc = (ax == 0) ? cx : ((ax == 1) ? cy : cz);
            relc[r][ax] = p[ax] - cc;
        }
        __syncthreads();
        // build gin chunk (16 x 160) in f16
        for (int i = tid; i < 16 * DINP; i += 256) {
            int r = i / DINP, c = i % DINP;
            float v;
            if (c < STEM) {
                v = feat[((long)b * N_ + nidxs[r]) * STEM + c];
            } else if (c < STEM + 3) {
                v = relc[r][c - STEM];
            } else if (c < DIN) {
                int q  = c - (STEM + 3);
                int ax = q >> 3, wi = q & 7;
                float fr = expf(-lg * (float)(wi & 3) / 3.0f);
                float a  = relc[r][ax] * fr;
                v = (wi < 4) ? sinf(a) : cosf(a);
            } else {
                v = 0.0f;
            }
            ginh[r][c] = (_Float16)v;
        }
        __syncthreads();
        // MLP1: each wave owns n-tiles {w, w+8}; K = 160 -> 5 WMMA steps
#pragma unroll
        for (int t = 0; t < 2; ++t) {
            int nt = w + t * 8;
            v8f acc = {};
#pragma unroll
            for (int ks = 0; ks < DINP / 32; ++ks) {
                v16h a  = load_a_frag(&ginh[0][0], DINP, ks * 32);
                v16h bb = load_b_frag_sw(W1, ks * 16 + nt);
                acc = __builtin_amdgcn_wmma_f32_16x16x32_f16(false, a, false, bb,
                                                             (short)0, acc, false, false);
            }
            int col   = nt * 16 + (lane & 15);
            int rbase = (lane < 16) ? 0 : 8;
#pragma unroll
            for (int i = 0; i < 8; ++i) raw[rbase + i][col] = acc[i] + b1s[col];
        }
        __syncthreads();
        // LayerNorm + GELU per row -> h1 (f16)
#pragma unroll
        for (int rr = 0; rr < 2; ++rr) {
            int r = w * 2 + rr;
            float s = 0.f, ss = 0.f, vals[8];
#pragma unroll
            for (int e = 0; e < 8; ++e) {
                float v = raw[r][lane + e * 32];
                vals[e] = v; s += v; ss += v * v;
            }
            s = wave_sum(s); ss = wave_sum(ss);
            float mean = s * (1.0f / TOK);
            float inv  = rsqrtf(ss * (1.0f / TOK) - mean * mean + EPS_);
#pragma unroll
            for (int e = 0; e < 8; ++e) {
                int c = lane + e * 32;
                float y = (vals[e] - mean) * inv * g1s[c] + be1s[c];
                h1[r][c] = (_Float16)gelu_exact(y);
            }
        }
        __syncthreads();
        // MLP2: K = 256 -> 8 WMMA steps
#pragma unroll
        for (int t = 0; t < 2; ++t) {
            int nt = w + t * 8;
            v8f acc = {};
#pragma unroll
            for (int ks = 0; ks < TOK / 32; ++ks) {
                v16h a  = load_a_frag(&h1[0][0], TOK, ks * 32);
                v16h bb = load_b_frag_sw(W2, ks * 16 + nt);
                acc = __builtin_amdgcn_wmma_f32_16x16x32_f16(false, a, false, bb,
                                                             (short)0, acc, false, false);
            }
            int col   = nt * 16 + (lane & 15);
            int rbase = (lane < 16) ? 0 : 8;
#pragma unroll
            for (int i = 0; i < 8; ++i) raw[rbase + i][col] = acc[i] + b2s[col];
        }
        __syncthreads();
        // LayerNorm (no act) in place
#pragma unroll
        for (int rr = 0; rr < 2; ++rr) {
            int r = w * 2 + rr;
            float s = 0.f, ss = 0.f, vals[8];
#pragma unroll
            for (int e = 0; e < 8; ++e) {
                float v = raw[r][lane + e * 32];
                vals[e] = v; s += v; ss += v * v;
            }
            s = wave_sum(s); ss = wave_sum(ss);
            float mean = s * (1.0f / TOK);
            float inv  = rsqrtf(ss * (1.0f / TOK) - mean * mean + EPS_);
#pragma unroll
            for (int e = 0; e < 8; ++e) {
                int c = lane + e * 32;
                raw[r][c] = (vals[e] - mean) * inv * g2s[c] + be2s[c];
            }
        }
        __syncthreads();
        // running max over neighbor rows
        float m = pooled[tid];
#pragma unroll
        for (int r = 0; r < 16; ++r) m = fmaxf(m, raw[r][tid]);
        pooled[tid] = m;
    }
    __syncthreads();
    tok_in[(long)cg * DTOK + STEM + scale * TOK + tid] = (_Float16)pooled[tid];
}

// ---------------- token MLP (992 -> 256 -> 256) with WMMA ----------------

__global__ void __launch_bounds__(256) token_mlp_kernel(
    const _Float16* __restrict__ tok_in,
    const _Float16* __restrict__ tW1sw, const _Float16* __restrict__ tW2sw,
    const float* __restrict__ tb1, const float* __restrict__ tg1, const float* __restrict__ tbe1,
    const float* __restrict__ tb2, const float* __restrict__ tg2, const float* __restrict__ tbe2,
    float* __restrict__ out_tok) {
    __shared__ float b1s[TOK], g1s[TOK], be1s[TOK], b2s[TOK], g2s[TOK], be2s[TOK];
    __shared__ float    raw[16][TOK];
    __shared__ _Float16 h1[16][TOK];
    int tid = threadIdx.x, w = tid >> 5, lane = tid & 31;
    int row0 = blockIdx.x * 16;
    b1s[tid] = tb1[tid]; g1s[tid] = tg1[tid]; be1s[tid] = tbe1[tid];
    b2s[tid] = tb2[tid]; g2s[tid] = tg2[tid]; be2s[tid] = tbe2[tid];
    __syncthreads();
    const _Float16* A = tok_in + (long)row0 * DTOK;
    // MLP1: K = 992 -> 31 WMMA steps
#pragma unroll 1
    for (int t = 0; t < 2; ++t) {
        int nt = w + t * 8;
        v8f acc = {};
        for (int ks = 0; ks < DTOK / 32; ++ks) {
            if (ks + 2 < DTOK / 32)
                __builtin_prefetch(tW1sw + ((long)((ks + 2) * 16 + nt) * 32) * 16, 0, 1);
            v16h a  = load_a_frag(A, DTOK, ks * 32);
            v16h bb = load_b_frag_sw(tW1sw, ks * 16 + nt);
            acc = __builtin_amdgcn_wmma_f32_16x16x32_f16(false, a, false, bb,
                                                         (short)0, acc, false, false);
        }
        int col   = nt * 16 + (lane & 15);
        int rbase = (lane < 16) ? 0 : 8;
#pragma unroll
        for (int i = 0; i < 8; ++i) raw[rbase + i][col] = acc[i] + b1s[col];
    }
    __syncthreads();
#pragma unroll
    for (int rr = 0; rr < 2; ++rr) {
        int r = w * 2 + rr;
        float s = 0.f, ss = 0.f, vals[8];
#pragma unroll
        for (int e = 0; e < 8; ++e) {
            float v = raw[r][lane + e * 32];
            vals[e] = v; s += v; ss += v * v;
        }
        s = wave_sum(s); ss = wave_sum(ss);
        float mean = s * (1.0f / TOK);
        float inv  = rsqrtf(ss * (1.0f / TOK) - mean * mean + EPS_);
#pragma unroll
        for (int e = 0; e < 8; ++e) {
            int c = lane + e * 32;
            float y = (vals[e] - mean) * inv * g1s[c] + be1s[c];
            h1[r][c] = (_Float16)gelu_exact(y);
        }
    }
    __syncthreads();
    // MLP2: K = 256 -> 8 WMMA steps
#pragma unroll
    for (int t = 0; t < 2; ++t) {
        int nt = w + t * 8;
        v8f acc = {};
#pragma unroll
        for (int ks = 0; ks < TOK / 32; ++ks) {
            v16h a  = load_a_frag(&h1[0][0], TOK, ks * 32);
            v16h bb = load_b_frag_sw(tW2sw, ks * 16 + nt);
            acc = __builtin_amdgcn_wmma_f32_16x16x32_f16(false, a, false, bb,
                                                         (short)0, acc, false, false);
        }
        int col   = nt * 16 + (lane & 15);
        int rbase = (lane < 16) ? 0 : 8;
#pragma unroll
        for (int i = 0; i < 8; ++i) raw[rbase + i][col] = acc[i] + b2s[col];
    }
    __syncthreads();
    // final LayerNorm, write f32 output
#pragma unroll
    for (int rr = 0; rr < 2; ++rr) {
        int r = w * 2 + rr;
        float s = 0.f, ss = 0.f, vals[8];
#pragma unroll
        for (int e = 0; e < 8; ++e) {
            float v = raw[r][lane + e * 32];
            vals[e] = v; s += v; ss += v * v;
        }
        s = wave_sum(s); ss = wave_sum(ss);
        float mean = s * (1.0f / TOK);
        float inv  = rsqrtf(ss * (1.0f / TOK) - mean * mean + EPS_);
#pragma unroll
        for (int e = 0; e < 8; ++e) {
            int c = lane + e * 32;
            out_tok[(long)(row0 + r) * TOK + c] = (vals[e] - mean) * inv * g2s[c] + be2s[c];
        }
    }
}

// ---------------- launch ----------------

extern "C" void kernel_launch(void* const* d_in, const int* in_sizes, int n_in,
                              void* d_out, int out_size, void* d_ws, size_t ws_size,
                              hipStream_t stream) {
    const float* xyz  = (const float*)d_in[0];
    const float* feat = (const float*)d_in[1];
    const float* sW1  = (const float*)d_in[2];
    const float* sb1  = (const float*)d_in[3];
    const float* sg1  = (const float*)d_in[4];
    const float* sbe1 = (const float*)d_in[5];
    const float* sW2  = (const float*)d_in[6];
    const float* sb2  = (const float*)d_in[7];
    const float* sg2  = (const float*)d_in[8];
    const float* sbe2 = (const float*)d_in[9];
    const float* tW1  = (const float*)d_in[10];
    const float* tb1  = (const float*)d_in[11];
    const float* tg1  = (const float*)d_in[12];
    const float* tbe1 = (const float*)d_in[13];
    const float* tW2  = (const float*)d_in[14];
    const float* tb2  = (const float*)d_in[15];
    const float* tg2  = (const float*)d_in[16];
    const float* tbe2 = (const float*)d_in[17];

    char* ws = (char*)d_ws;
    size_t off = 0;
    auto alloc = [&](size_t bytes) {
        size_t o = off;
        off = (off + bytes + 255) & ~(size_t)255;
        return o;
    };
    int*      center_idx   = (int*)     (ws + alloc((size_t)B_ * S_ * sizeof(int)));
    float*    patch_center = (float*)   (ws + alloc((size_t)B_ * S_ * 3 * sizeof(float)));
    int*      nbr          = (int*)     (ws + alloc((size_t)3 * B_ * S_ * 64 * sizeof(int)));
    _Float16* sW1sw        = (_Float16*)(ws + alloc((size_t)3 * DINP * TOK * sizeof(_Float16)));
    _Float16* sW2sw        = (_Float16*)(ws + alloc((size_t)3 * TOK * TOK * sizeof(_Float16)));
    _Float16* tW1sw        = (_Float16*)(ws + alloc((size_t)DTOK * TOK * sizeof(_Float16)));
    _Float16* tW2sw        = (_Float16*)(ws + alloc((size_t)TOK * TOK * sizeof(_Float16)));
    _Float16* tok_in       = (_Float16*)(ws + alloc((size_t)B_ * S_ * DTOK * sizeof(_Float16)));

    float* out_tok = (float*)d_out;
    float* out_pc  = out_tok + (size_t)B_ * S_ * TOK;

    // 1) weight conversion to f16, K padding + B-fragment swizzle
    convert_swizzle_kernel<<<256, 256, 0, stream>>>(sW1, sW1sw, DIN, DINP, 3);
    convert_swizzle_kernel<<<256, 256, 0, stream>>>(sW2, sW2sw, TOK, TOK, 3);
    convert_swizzle_kernel<<<256, 256, 0, stream>>>(tW1, tW1sw, DTOK, DTOK, 1);
    convert_swizzle_kernel<<<64, 256, 0, stream>>>(tW2, tW2sw, TOK, TOK, 1);

    // 2) farthest point sampling: one workgroup per batch
    fps_kernel<<<B_, 256, 0, stream>>>(xyz, center_idx);

    // 3) gather centers, center PE, patch_center output
    gather_centers_kernel<<<(B_ * S_ + 255) / 256, 256, 0, stream>>>(
        xyz, feat, center_idx, patch_center, tok_in, out_pc);

    // 4) ball query: one wave32 per center, 3 scales
    ball_query_kernel<<<dim3((B_ * S_) / 8, 3), 256, 0, stream>>>(xyz, patch_center, nbr);

    // 5) per-scale group MLP with WMMA + max pool
    group_mlp_kernel<<<dim3(B_ * S_, 3), 256, 0, stream>>>(
        xyz, feat, patch_center, nbr,
        sb1, sg1, sbe1, sb2, sg2, sbe2, sW1sw, sW2sw, tok_in);

    // 6) token MLP with WMMA, f32 output
    token_mlp_kernel<<<(B_ * S_) / 16, 256, 0, stream>>>(
        tok_in, tW1sw, tW2sw, tb1, tg1, tbe1, tb2, tg2, tbe2, out_tok);

    (void)in_sizes; (void)n_in; (void)out_size; (void)ws_size;
}
